// MultiHeadAttention_54674933678316
// MI455X (gfx1250) — matrix operand order repaired and checked by reading, not compiled, not run on hardware
//
#include <hip/hip_runtime.h>
#include <hip/hip_bf16.h>

// MI455X / gfx1250 MHA forward. wave32, WMMA f32_16x16x32_f16 everywhere.
// B=4, T=2048, D=1024, H=16, DK=64.
// Projections: LDS-staged (async-to-LDS, double buffered) weight tiles.

typedef _Float16 h16;
typedef __attribute__((ext_vector_type(16))) _Float16 v16h;
typedef __attribute__((ext_vector_type(8)))  _Float16 v8h;
typedef __attribute__((ext_vector_type(8)))  float    v8f;
typedef __attribute__((ext_vector_type(4)))  float    v4f;
typedef __attribute__((ext_vector_type(4)))  int      v4i;

#define TT 2048
#define DD 1024
#define HH 16
#define DK 64

#if __has_builtin(__builtin_amdgcn_global_load_async_to_lds_b128)
#define HAVE_ASYNC_LDS 1
#else
#define HAVE_ASYNC_LDS 0
#endif

__device__ __forceinline__ void async_copy16(const h16* src, h16* dst_lds) {
#if HAVE_ASYNC_LDS
  // (AS1 v4i*, AS3 v4i*, imm offset, imm cpol)
  __builtin_amdgcn_global_load_async_to_lds_b128(
      (__attribute__((address_space(1))) v4i*)src,
      (__attribute__((address_space(3))) v4i*)dst_lds, 0, 0);
#else
  *(v8h*)dst_lds = *(const v8h*)src;
#endif
}

__device__ __forceinline__ void async_wait_all() {
#if HAVE_ASYNC_LDS
#if __has_builtin(__builtin_amdgcn_s_wait_asynccnt)
  __builtin_amdgcn_s_wait_asynccnt(0);
#else
  asm volatile("s_wait_asynccnt 0x0" ::: "memory");
#endif
#endif
}

__device__ __forceinline__ v8f vzero8() {
  v8f z;
#pragma unroll
  for (int i = 0; i < 8; ++i) z[i] = 0.0f;
  return z;
}

__device__ __forceinline__ v8f wmma_f(v16h a, v16h b, v8f c) {
  return __builtin_amdgcn_wmma_f32_16x16x32_f16(false, a, false, b, (short)0, c,
                                                false, false);
}

// A-frag (16x32) from row-major f16. p = row base + k0. Lane: m = lane&15.
// halves 0..15 <-> K = k0 + hi*16 + 0..15
__device__ __forceinline__ v16h load_a_h(const h16* p, int hi) {
  v8h lo = *(const v8h*)(p + hi * 16);
  v8h hh = *(const v8h*)(p + hi * 16 + 8);
  v16h a;
#pragma unroll
  for (int i = 0; i < 8; ++i) { a[i] = lo[i]; a[i + 8] = hh[i]; }
  return a;
}

// Same A-frag from fp32 source, converted inline (v_cvt_pk_f16_f32).
__device__ __forceinline__ v16h load_a_f32(const float* p, int hi) {
  v4f f0 = *(const v4f*)(p + hi * 16);
  v4f f1 = *(const v4f*)(p + hi * 16 + 4);
  v4f f2 = *(const v4f*)(p + hi * 16 + 8);
  v4f f3 = *(const v4f*)(p + hi * 16 + 12);
  v16h a;
#pragma unroll
  for (int i = 0; i < 4; ++i) {
    a[i]      = (h16)f0[i];
    a[i + 4]  = (h16)f1[i];
    a[i + 8]  = (h16)f2[i];
    a[i + 12] = (h16)f3[i];
  }
  return a;
}

// B-frag (32x16): lane column n = lane&15; halves <-> K = hi*16 + 0..15.
// p points at the 16 contiguous K values for this lane's column.
__device__ __forceinline__ v16h load_b(const h16* p) {
  v8h b0 = *(const v8h*)(p);
  v8h b1 = *(const v8h*)(p + 8);
  v16h b;
#pragma unroll
  for (int i = 0; i < 8; ++i) { b[i] = b0[i]; b[i + 8] = b1[i]; }
  return b;
}

__global__ void cvt_f32_f16(const float* __restrict__ src, h16* __restrict__ dst, int n) {
  int i = blockIdx.x * blockDim.x + threadIdx.x;
  if (i < n) dst[i] = (h16)src[i];
}

// Cooperative stage of one 64(n) x 32(k) f16 weight tile into LDS.
// 256 x 16B chunks, 2 per thread (blockDim 128).
__device__ __forceinline__ void stage_btile(const h16* __restrict__ W, int n0,
                                            int k0, h16* buf, int tid) {
#pragma unroll
  for (int c = 0; c < 2; ++c) {
    const int chunk = tid + c * 128;
    const int row = chunk >> 2;   // 0..63  (n within tile)
    const int part = chunk & 3;   // 16B piece of the 64B row
    async_copy16(W + (size_t)(n0 + row) * DD + k0 + part * 8,
                 buf + row * 32 + part * 8);
  }
}

// C = X(fp32, M x 1024) @ W^T(f16, W is [n][k] row-major) + bias, store f16.
// mode 0: head-major dst[((b*H+h)*T + t)*64 + d]   (Q, K)
// mode 1: d-major    dst[((b*H+h)*64 + d)*T + t]   (V transposed)
__global__ __launch_bounds__(128)
void proj_qkv_kernel(const float* __restrict__ X, const h16* __restrict__ W,
                     const float* __restrict__ bias, h16* __restrict__ dst, int mode) {
  const int tid  = threadIdx.x;
  const int lane = tid & 31;
  const int wid  = tid >> 5;
  const int n16  = lane & 15;
  const int hi   = lane >> 4;
  const int m0   = (blockIdx.x * 4 + wid) * 16;
  const int n0   = blockIdx.y * 64;

  __shared__ __align__(16) h16 bsh[2][64 * 32];  // double-buffered B tile, 8KB

  stage_btile(W, n0, 0, &bsh[0][0], tid);

  v8f acc0 = vzero8(), acc1 = vzero8(), acc2 = vzero8(), acc3 = vzero8();
  const float* arow = X + (size_t)(m0 + n16) * DD;
  v16h a = load_a_f32(arow, hi);  // prefetched A-frag for k0 = 0

  for (int k0 = 0; k0 < DD; k0 += 32) {
    const int buf = (k0 >> 5) & 1;
    async_wait_all();
    __syncthreads();
    if (k0 + 32 < DD) stage_btile(W, n0, k0 + 32, &bsh[buf ^ 1][0], tid);
    const v16h acur = a;
    if (k0 + 32 < DD) a = load_a_f32(arow + k0 + 32, hi);  // prefetch next A
    // Materialize all 4 B-frags first: ds_load clause, staggered waits.
    const h16* bb = &bsh[buf][n16 * 32 + hi * 16];
    const v16h b0 = load_b(bb);
    const v16h b1 = load_b(bb + 16 * 32);
    const v16h b2 = load_b(bb + 32 * 32);
    const v16h b3 = load_b(bb + 48 * 32);
    acc0 = wmma_f(acur, b0, acc0);
    acc1 = wmma_f(acur, b1, acc1);
    acc2 = wmma_f(acur, b2, acc2);
    acc3 = wmma_f(acur, b3, acc3);
  }

  const int b     = m0 >> 11;        // T = 2048 rows per batch
  const int tbase = m0 & (TT - 1);
  const int h     = n0 >> 6;         // 64-wide tile stays inside one head
  v8f accs[4] = {acc0, acc1, acc2, acc3};
#pragma unroll
  for (int nt = 0; nt < 4; ++nt) {
    const int d = nt * 16 + n16;
    const float bv = bias[n0 + d];
#pragma unroll
    for (int r = 0; r < 8; ++r) {
      const int t = tbase + r + 8 * hi;
      const float val = accs[nt][r] + bv;
      size_t idx;
      if (mode == 0) idx = ((size_t)(b * HH + h) * TT + t) * DK + d;
      else           idx = ((size_t)(b * HH + h) * DK + d) * TT + t;
      dst[idx] = (h16)val;
    }
  }
}

// Final projection: out(fp32) = Of(f16) @ w_o^T + b_o
__global__ __launch_bounds__(128)
void proj_out_kernel(const h16* __restrict__ A, const h16* __restrict__ W,
                     const float* __restrict__ bias, float* __restrict__ out) {
  const int tid  = threadIdx.x;
  const int lane = tid & 31;
  const int wid  = tid >> 5;
  const int n16  = lane & 15;
  const int hi   = lane >> 4;
  const int m0   = (blockIdx.x * 4 + wid) * 16;
  const int n0   = blockIdx.y * 64;

  __shared__ __align__(16) h16 bsh[2][64 * 32];

  stage_btile(W, n0, 0, &bsh[0][0], tid);

  v8f acc0 = vzero8(), acc1 = vzero8(), acc2 = vzero8(), acc3 = vzero8();
  const h16* arow = A + (size_t)(m0 + n16) * DD;
  v16h a = load_a_h(arow, hi);

  for (int k0 = 0; k0 < DD; k0 += 32) {
    const int buf = (k0 >> 5) & 1;
    async_wait_all();
    __syncthreads();
    if (k0 + 32 < DD) stage_btile(W, n0, k0 + 32, &bsh[buf ^ 1][0], tid);
    const v16h acur = a;
    if (k0 + 32 < DD) a = load_a_h(arow + k0 + 32, hi);
    const h16* bb = &bsh[buf][n16 * 32 + hi * 16];
    const v16h b0 = load_b(bb);
    const v16h b1 = load_b(bb + 16 * 32);
    const v16h b2 = load_b(bb + 32 * 32);
    const v16h b3 = load_b(bb + 48 * 32);
    acc0 = wmma_f(acur, b0, acc0);
    acc1 = wmma_f(acur, b1, acc1);
    acc2 = wmma_f(acur, b2, acc2);
    acc3 = wmma_f(acur, b3, acc3);
  }

  v8f accs[4] = {acc0, acc1, acc2, acc3};
#pragma unroll
  for (int nt = 0; nt < 4; ++nt) {
    const int n = n0 + nt * 16 + n16;
    const float bv = bias[n];
#pragma unroll
    for (int r = 0; r < 8; ++r)
      out[(size_t)(m0 + r + 8 * hi) * DD + n] = accs[nt][r] + bv;
  }
}

// Flash attention, one wave per (b,h, 16-query-row tile). Causal.
// Qh/Kh: head-major [bh][t][d] f16.  Vt: d-major [bh][d][t] f16.
// Of: [b*T + t][h*64 + d] f16 (row-major over D=1024 for the final GEMM).
__global__ __launch_bounds__(32)
void attn_kernel(const h16* __restrict__ Qh, const h16* __restrict__ Kh,
                 const h16* __restrict__ Vt, h16* __restrict__ Of) {
  const int lane = threadIdx.x & 31;
  const int n16  = lane & 15;
  const int hi   = lane >> 4;
  const int qb   = blockIdx.x * 16;
  const int bh   = blockIdx.y;

  const h16* qbase = Qh + (size_t)bh * TT * DK;
  const h16* kbase = Kh + (size_t)bh * TT * DK;
  const h16* vbase = Vt + (size_t)bh * DK * TT;

  const h16* qrowp = qbase + (size_t)(qb + n16) * DK;
  const v16h qf0 = load_a_h(qrowp + 0,  hi);
  const v16h qf1 = load_a_h(qrowp + 32, hi);

  float mi[8], li[8];
#pragma unroll
  for (int r = 0; r < 8; ++r) { mi[r] = -1e30f; li[r] = 0.0f; }
  v8f o0 = vzero8(), o1 = vzero8(), o2 = vzero8(), o3 = vzero8();

  __shared__ __align__(16) h16 plds[16 * 32];

  const int nkt = (qb + 16 + 31) >> 5;  // 32-key tiles up to the diagonal
  for (int j = 0; j < nkt; ++j) {
    const int kb = j * 32;
    // Materialize all 4 K B-frags (load clause), then the 4 score WMMAs.
    const h16* kp0 = kbase + (size_t)(kb + n16) * DK + hi * 16;
    const h16* kp1 = kbase + (size_t)(kb + 16 + n16) * DK + hi * 16;
    const v16h bk00 = load_b(kp0);
    const v16h bk01 = load_b(kp0 + 32);
    const v16h bk10 = load_b(kp1);
    const v16h bk11 = load_b(kp1 + 32);
    v8f s0 = vzero8(), s1 = vzero8();
    s0 = wmma_f(qf0, bk00, s0);
    s0 = wmma_f(qf1, bk01, s0);
    s1 = wmma_f(qf0, bk10, s1);
    s1 = wmma_f(qf1, bk11, s1);

    // Hoist V B-frags: overlap these global loads with the softmax VALU work.
    const h16* vp = vbase + (size_t)n16 * TT + kb + hi * 16;
    const v16h bv0 = load_b(vp);
    const v16h bv1 = load_b(vp + 16 * TT);
    const v16h bv2 = load_b(vp + 32 * TT);
    const v16h bv3 = load_b(vp + 48 * TT);

    float alpha[8];
#pragma unroll
    for (int r = 0; r < 8; ++r) {
      const int qrow = qb + r + 8 * hi;             // C-layout row for slot r
      float x0 = (kb + n16      <= qrow) ? s0[r] * 0.125f : -1e30f;
      float x1 = (kb + 16 + n16 <= qrow) ? s1[r] * 0.125f : -1e30f;
      float mx = fmaxf(x0, x1);
      mx = fmaxf(mx, __shfl_xor(mx, 1, 32));
      mx = fmaxf(mx, __shfl_xor(mx, 2, 32));
      mx = fmaxf(mx, __shfl_xor(mx, 4, 32));
      mx = fmaxf(mx, __shfl_xor(mx, 8, 32));
      const float mnew = fmaxf(mi[r], mx);
      const float aa = __expf(mi[r] - mnew);
      const float p0 = __expf(x0 - mnew);
      const float p1 = __expf(x1 - mnew);
      float rs = p0 + p1;
      rs += __shfl_xor(rs, 1, 32);
      rs += __shfl_xor(rs, 2, 32);
      rs += __shfl_xor(rs, 4, 32);
      rs += __shfl_xor(rs, 8, 32);
      li[r] = li[r] * aa + rs;
      mi[r] = mnew;
      alpha[r] = aa;
      plds[(r + 8 * hi) * 32 + n16]      = (h16)p0;
      plds[(r + 8 * hi) * 32 + 16 + n16] = (h16)p1;
    }
#pragma unroll
    for (int r = 0; r < 8; ++r) {
      o0[r] *= alpha[r]; o1[r] *= alpha[r]; o2[r] *= alpha[r]; o3[r] *= alpha[r];
    }
    __syncthreads();  // C-layout -> A-layout reshuffle of P through LDS
    const v16h pf = load_a_h(plds + n16 * 32, hi);
    o0 = wmma_f(pf, bv0, o0);
    o1 = wmma_f(pf, bv1, o1);
    o2 = wmma_f(pf, bv2, o2);
    o3 = wmma_f(pf, bv3, o3);
    __syncthreads();
  }

  const int b = bh >> 4, h = bh & 15;
#pragma unroll
  for (int r = 0; r < 8; ++r) {
    const float inv = 1.0f / li[r];
    const int t = qb + r + 8 * hi;
    const size_t base = ((size_t)(b * TT + t)) * DD + h * DK;
    Of[base + n16]      = (h16)(o0[r] * inv);
    Of[base + 16 + n16] = (h16)(o1[r] * inv);
    Of[base + 32 + n16] = (h16)(o2[r] * inv);
    Of[base + 48 + n16] = (h16)(o3[r] * inv);
  }
}

extern "C" void kernel_launch(void* const* d_in, const int* in_sizes, int n_in,
                              void* d_out, int out_size, void* d_ws, size_t ws_size,
                              hipStream_t stream) {
  const float* q   = (const float*)d_in[0];
  const float* k   = (const float*)d_in[1];
  const float* v   = (const float*)d_in[2];
  // d_in[3]: causal mask — implemented analytically in attn_kernel
  const float* w_q = (const float*)d_in[4];
  const float* b_q = (const float*)d_in[5];
  const float* w_k = (const float*)d_in[6];
  const float* b_k = (const float*)d_in[7];
  const float* w_v = (const float*)d_in[8];
  const float* b_v = (const float*)d_in[9];
  const float* w_o = (const float*)d_in[10];
  const float* b_o = (const float*)d_in[11];

  char* ws = (char*)d_ws;                      // 72 MB total
  h16* wq16 = (h16*)(ws + (size_t)0);          // 2 MB each weight
  h16* wk16 = (h16*)(ws + ((size_t)2  << 20));
  h16* wv16 = (h16*)(ws + ((size_t)4  << 20));
  h16* wo16 = (h16*)(ws + ((size_t)6  << 20));
  h16* qh   = (h16*)(ws + ((size_t)8  << 20)); // 16 MB each activation
  h16* kh   = (h16*)(ws + ((size_t)24 << 20));
  h16* vt   = (h16*)(ws + ((size_t)40 << 20));
  h16* of   = (h16*)(ws + ((size_t)56 << 20));

  const int NW = DD * DD;
  cvt_f32_f16<<<NW / 256, 256, 0, stream>>>(w_q, wq16, NW);
  cvt_f32_f16<<<NW / 256, 256, 0, stream>>>(w_k, wk16, NW);
  cvt_f32_f16<<<NW / 256, 256, 0, stream>>>(w_v, wv16, NW);
  cvt_f32_f16<<<NW / 256, 256, 0, stream>>>(w_o, wo16, NW);

  dim3 pgrid(128, 16);  // (8192/16)/4 m-strips x (1024/64) n-tiles
  proj_qkv_kernel<<<pgrid, 128, 0, stream>>>(q, wq16, b_q, qh, 0);
  proj_qkv_kernel<<<pgrid, 128, 0, stream>>>(k, wk16, b_k, kh, 0);
  proj_qkv_kernel<<<pgrid, 128, 0, stream>>>(v, wv16, b_v, vt, 1);

  attn_kernel<<<dim3(TT / 16, 4 * HH), 32, 0, stream>>>(qh, kh, vt, of);

  proj_out_kernel<<<pgrid, 128, 0, stream>>>(of, wo16, b_o, (float*)d_out);
}
